// AnchoredSubgraphRAG_15985868275721
// MI455X (gfx1250) — matrix operand order repaired and checked by reading, not compiled, not run on hardware
//
#include <hip/hip_runtime.h>
#include <stdint.h>

// ---------------------------------------------------------------------------
// Types for CDNA5 WMMA (wave32, 16x16x32 bf16 -> f32)
// ---------------------------------------------------------------------------
typedef __attribute__((ext_vector_type(16))) __bf16 v16bf;
typedef __attribute__((ext_vector_type(8)))  float  v8f;

#if defined(__has_builtin)
#if __has_builtin(__builtin_amdgcn_global_load_async_to_lds_b128)
#define HAVE_ASYNC_LDS 1
#endif
#endif

#ifdef HAVE_ASYNC_LDS
// exact parameter type per compiler diagnostic: vector_size(16) int, AS1/AS3
typedef int v4i_vs __attribute__((vector_size(16)));
#define ASYNC_G(p) ((__attribute__((address_space(1))) v4i_vs*)(p))
#define ASYNC_L(p) ((__attribute__((address_space(3))) v4i_vs*)(p))
#endif

__device__ __forceinline__ unsigned short f2bf(float f) {
  unsigned int u = __float_as_uint(f);
  u += 0x7FFFu + ((u >> 16) & 1u);   // round-to-nearest-even
  return (unsigned short)(u >> 16);
}

// ---------------------------------------------------------------------------
// Generic tiled GEMM: C[M,256] = A[M,K](f32,row-major,lda) * B[K,256]
// B supplied pre-transposed + bf16-converted: BT[256][K] (ushort bf16 bits).
// Tile 128x128x32; 8 waves (4 in M, 2 in N); each wave 2x4 16x16 frags.
// B tile staged via GLOBAL_LOAD_ASYNC_TO_LDS_B128 (ASYNCcnt) when available.
// epi: 0 none | 1 relu(acc+bias[col]) | 2 sigmoid(acc + C[off]) (accumulate)
//      | 3 ex1[off] + ex2[off]*tanh(acc)
// ---------------------------------------------------------------------------
__global__ __launch_bounds__(256) void gemm_bf16_wmma(
    const float* __restrict__ A, int lda,
    const unsigned short* __restrict__ BT, int K,
    float* __restrict__ C, int M,
    int epi,
    const float* __restrict__ bias,
    const float* __restrict__ ex1,
    const float* __restrict__ ex2)
{
  __shared__ __align__(16) unsigned short As[128 * 32];
  __shared__ __align__(16) unsigned short Bs[128 * 32];

  const int tid  = threadIdx.x;
  const int wave = tid >> 5;
  const int lane = tid & 31;
  const int wm = wave & 3;      // 0..3 : 32-row strip
  const int wn = wave >> 2;     // 0..1 : 64-col strip
  const int m0 = blockIdx.x * 128;
  const int n0 = blockIdx.y * 128;

  v8f acc[2][4];
  const v8f vzero = {0.f,0.f,0.f,0.f,0.f,0.f,0.f,0.f};
#pragma unroll
  for (int i = 0; i < 2; i++)
#pragma unroll
    for (int j = 0; j < 4; j++) acc[i][j] = vzero;

  const int hsel = lane >> 4;   // 0: lanes 0-15, 1: lanes 16-31
  const int l16  = lane & 15;

  for (int k0 = 0; k0 < K; k0 += 32) {
    // ---- stage B tile via async global->LDS DMA (pure bf16 copy, no VALU)
    {
      int n  = tid >> 1;
      int cp = (tid & 1) * 16;
      const unsigned short* gp = BT + (size_t)(n0 + n) * K + k0 + cp;
      unsigned short* lp = &Bs[n * 32 + cp];
#ifdef HAVE_ASYNC_LDS
      __builtin_amdgcn_global_load_async_to_lds_b128(ASYNC_G(gp),     ASYNC_L(lp),     0, 0);
      __builtin_amdgcn_global_load_async_to_lds_b128(ASYNC_G(gp + 8), ASYNC_L(lp + 8), 0, 0);
#else
      const uint4* bp = (const uint4*)gp;
      *(uint4*)lp       = bp[0];
      *(uint4*)(lp + 8) = bp[1];
#endif
    }
    // ---- stage A tile (convert f32 -> bf16 in registers) : 128 rows x 32 K
    {
      int r  = tid >> 1;
      int cp = (tid & 1) * 16;
      int gr = m0 + r;
      float4 f0, f1, f2, f3;
      if (gr < M) {
        const float4* ap = (const float4*)(A + (size_t)gr * lda + k0 + cp);
        f0 = ap[0]; f1 = ap[1]; f2 = ap[2]; f3 = ap[3];
        if (k0 + 32 < K)  // prefetch next k-step of this row toward the WGP
          __builtin_prefetch(A + (size_t)gr * lda + k0 + 32 + cp, 0, 1);
      } else {
        f0 = f1 = f2 = f3 = make_float4(0.f, 0.f, 0.f, 0.f);
      }
      unsigned short* d = &As[r * 32 + cp];
      d[0]=f2bf(f0.x); d[1]=f2bf(f0.y); d[2]=f2bf(f0.z); d[3]=f2bf(f0.w);
      d[4]=f2bf(f1.x); d[5]=f2bf(f1.y); d[6]=f2bf(f1.z); d[7]=f2bf(f1.w);
      d[8]=f2bf(f2.x); d[9]=f2bf(f2.y); d[10]=f2bf(f2.z); d[11]=f2bf(f2.w);
      d[12]=f2bf(f3.x); d[13]=f2bf(f3.y); d[14]=f2bf(f3.z); d[15]=f2bf(f3.w);
    }
#ifdef HAVE_ASYNC_LDS
#if __has_builtin(__builtin_amdgcn_s_wait_asynccnt)
    __builtin_amdgcn_s_wait_asynccnt(0);
#else
    asm volatile("s_wait_asynccnt 0" ::: "memory");
#endif
#endif
    __syncthreads();

    // ---- fragment loads per ISA wave32 layouts
    union Frag { v16bf v; uint4 q[2]; } a[2], b[4];
#pragma unroll
    for (int fm = 0; fm < 2; fm++) {
      const unsigned short* p = &As[(wm * 32 + fm * 16 + l16) * 32];
      a[fm].q[0] = *(const uint4*)(p + 8 * hsel);        // K j..j+7
      a[fm].q[1] = *(const uint4*)(p + 16 + 8 * hsel);   // K j+16..j+23
    }
#pragma unroll
    for (int fn = 0; fn < 4; fn++) {
      const unsigned short* p = &Bs[(wn * 64 + fn * 16 + l16) * 32 + 8 * hsel];
      b[fn].q[0] = *(const uint4*)(p);
      b[fn].q[1] = *(const uint4*)(p + 16);
    }
#pragma unroll
    for (int fm = 0; fm < 2; fm++)
#pragma unroll
      for (int fn = 0; fn < 4; fn++)
        acc[fm][fn] = __builtin_amdgcn_wmma_f32_16x16x32_bf16(
            false, a[fm].v, false, b[fn].v, (short)0, acc[fm][fn], false, false);
    __syncthreads();
  }

  // ---- epilogue (C/D layout: VGPR r -> row r + 8*hsel, col = lane&15)
#pragma unroll
  for (int fm = 0; fm < 2; fm++)
#pragma unroll
    for (int fn = 0; fn < 4; fn++)
#pragma unroll
      for (int r = 0; r < 8; r++) {
        int row = m0 + wm * 32 + fm * 16 + r + 8 * hsel;
        if (row >= M) continue;
        int col = n0 + wn * 64 + fn * 16 + l16;
        size_t off = (size_t)row * 256 + col;
        float v = acc[fm][fn][r];
        if (epi == 1)      v = fmaxf(v + bias[col], 0.f);
        else if (epi == 2) { v += C[off]; v = 1.f / (1.f + __expf(-v)); }
        else if (epi == 3) { v = ex1[off] + ex2[off] * tanhf(v); }
        C[off] = v;
      }
}

// ---------------------------------------------------------------------------
// Small utility kernels
// ---------------------------------------------------------------------------
__global__ void zero_f32(float* p, int n) {
  int i = blockIdx.x * 256 + threadIdx.x;
  if (i < n) p[i] = 0.f;
}
__global__ void copy_f32(const float* s, float* d, int n) {
  int i = blockIdx.x * 256 + threadIdx.x;
  if (i < n) d[i] = s[i];
}
__global__ void copy_i32(const int* s, int* d, int n) {
  int i = blockIdx.x * 256 + threadIdx.x;
  if (i < n) d[i] = s[i];
}

// weight convert+transpose: W[K][256] f32 -> WT[256][Kp] bf16 (zero K pad)
__global__ void wt_convert(const float* __restrict__ W, unsigned short* __restrict__ WT,
                           int K, int Kp) {
  int idx = blockIdx.x * 256 + threadIdx.x;
  if (idx >= 256 * Kp) return;
  int n = idx / Kp, k = idx - n * Kp;
  WT[(size_t)n * Kp + k] = (k < K) ? f2bf(W[(size_t)k * 256 + n]) : (unsigned short)0;
}

// dot of each row with q (wave per row, 256-dim), with non-text fallback row
__global__ void row_dot_q(const float* __restrict__ rows, const float* __restrict__ fallback,
                          int ncut, const float* __restrict__ q, float* __restrict__ out, int n) {
  int e = blockIdx.x * 8 + (threadIdx.x >> 5);
  int lane = threadIdx.x & 31;
  if (e >= n) return;
  const float* row = (e < ncut) ? rows + (size_t)e * 256 : fallback;
  float s = 0.f;
#pragma unroll
  for (int u = 0; u < 8; u++) { int c = lane + 32 * u; s += row[c] * q[c]; }
#pragma unroll
  for (int off = 16; off; off >>= 1) s += __shfl_xor(s, off, 32);
  if (lane == 0) out[e] = s;
}

__global__ void tri_score_kernel(const int* h, const int* r, const int* t,
                                 const float* qe, const float* qr, float* tri, int E) {
  int i = blockIdx.x * 256 + threadIdx.x;
  if (i < E) tri[i] = qe[h[i]] + qr[r[i]] + qe[t[i]];
}

// 32 sequential masked argmax passes, single block of 1024 threads
__global__ void topk32_kernel(float* tri, int E, int* aidx, float* ascore) {
  __shared__ float sv[1024];
  __shared__ int   si[1024];
  int t = threadIdx.x;
  for (int k = 0; k < 32; k++) {
    float best = -3.4e38f; int bi = 0x7fffffff;
    for (int i = t; i < E; i += 1024) {
      float v = tri[i];
      if (v > best || (v == best && i < bi)) { best = v; bi = i; }
    }
    sv[t] = best; si[t] = bi;
    __syncthreads();
    for (int s = 512; s > 0; s >>= 1) {
      if (t < s) {
        if (sv[t + s] > sv[t] || (sv[t + s] == sv[t] && si[t + s] < si[t])) {
          sv[t] = sv[t + s]; si[t] = si[t + s];
        }
      }
      __syncthreads();
    }
    if (t == 0) { aidx[k] = si[0]; ascore[k] = sv[0]; tri[si[0]] = -3.4e38f; }
    __syncthreads();
  }
}

__global__ void topic_scatter(const int* ids, float* topic, int n) {
  int i = threadIdx.x;
  if (i < n) topic[ids[i]] = 1.f;
}
__global__ void deg_kernel(const int* h, const int* t, float* degh, float* degt, int E) {
  int i = blockIdx.x * 256 + threadIdx.x;
  if (i >= E) return;
  atomicAdd(&degt[t[i]], 1.f);
  atomicAdd(&degh[h[i]], 1.f);
}
__global__ void anchor_scatter(const int* aidx, const float* ascore,
                               const int* h, const int* t,
                               float* mask, float* isA, float* aS) {
  int k = threadIdx.x;
  if (k >= 32) return;
  int e = aidx[k];
  mask[h[e]] = 1.f; mask[t[e]] = 1.f;
  isA[e] = 1.f; aS[e] = ascore[k];
}

// one DDE propagation hop: dst[di[e]] += src[si_[e]]
__global__ void dde_edge(const int* si_, const int* di, const float* src, float* dst, int E) {
  int i = blockIdx.x * 256 + threadIdx.x;
  if (i >= E) return;
  float v = src[si_[i]];
  if (v != 0.f) atomicAdd(&dst[di[i]], v);
}
__global__ void dde_div_store(float* s, const float* deg, float* pe, int col, int N) {
  int i = blockIdx.x * 256 + threadIdx.x;
  if (i >= N) return;
  float v = s[i] / fmaxf(deg[i], 1.f);
  s[i] = v;
  pe[(size_t)i * 4 + col] = v;
}

__global__ void bfs_init(const float* mask, int* dist, int N) {
  int i = blockIdx.x * 256 + threadIdx.x;
  if (i < N) dist[i] = (mask[i] > 0.f) ? 0 : 4;   // FAR = MAX_DIST+1 = 4
}
__global__ void bfs_edge(const int* h, const int* t, const int* dist, int* nd, int E) {
  int i = blockIdx.x * 256 + threadIdx.x;
  if (i >= E) return;
  atomicMin(&nd[t[i]], dist[h[i]] + 1);
  atomicMin(&nd[h[i]], dist[t[i]] + 1);
}

// assemble encoder input X[N][288] = [emb(256)|pe_topic(4)|dist_emb(16)|topic,mask,1|pad(9)]
__global__ void xenc_assemble(const float* __restrict__ ent, const float* __restrict__ nte,
                              int Ntext,
                              const float* __restrict__ pe_topic, const int* __restrict__ dist,
                              const float* __restrict__ dist_emb,
                              const float* __restrict__ topic, const float* __restrict__ mask,
                              float* __restrict__ X) {
  int i = blockIdx.x;
  int k = threadIdx.x;
  float v;
  if (k < 256)       v = (i < Ntext) ? ent[(size_t)i * 256 + k] : nte[k];
  else if (k < 260)  v = pe_topic[(size_t)i * 4 + (k - 256)];
  else if (k < 276)  v = dist_emb[dist[i] * 16 + (k - 260)];
  else if (k == 276) v = topic[i];
  else if (k == 277) v = mask[i];
  else if (k == 278) v = 1.f;
  else               v = 0.f;
  X[(size_t)i * 288 + k] = v;
}

// per-edge message: agg[t] += relu(Am[h] + Bm[r])   (wave per edge)
__global__ void edge_msg(const int* h_ids, const int* r_ids, const int* t_ids,
                         const float* __restrict__ Am, const float* __restrict__ Bm,
                         float* __restrict__ agg, int E) {
  int e = blockIdx.x * 8 + (threadIdx.x >> 5);
  int lane = threadIdx.x & 31;
  if (e >= E) return;
  int h = h_ids[e], r = r_ids[e], t = t_ids[e];
  const float* am = Am + (size_t)h * 256;
  const float* bm = Bm + (size_t)r * 256;
  float* ag = agg + (size_t)t * 256;
#pragma unroll
  for (int u = 0; u < 8; u++) {
    int c = lane + 32 * u;
    float v = fmaxf(am[c] + bm[c], 0.f);
    atomicAdd(&ag[c], v);
  }
}

// q_contrib[c] = sum_k q[k] * W1[k][c]  (W1 rows 0..255)
__global__ void q_contrib_kernel(const float* q, const float* W1, float* qc) {
  int c = threadIdx.x;
  float s = 0.f;
  for (int k = 0; k < 256; k++) s += q[k] * W1[(size_t)k * 256 + c];
  qc[c] = s;
}

// final scorer: wave per edge; hid = relu(sum of precomputed parts); out = hid . W2 + b2
__global__ void scorer_edge(const int* h_ids, const int* r_ids, const int* t_ids,
                            const float* __restrict__ preH, const float* __restrict__ preR,
                            const float* __restrict__ preT,
                            const float* __restrict__ qc, const float* __restrict__ b1,
                            const float* __restrict__ isA, const float* __restrict__ aS,
                            const float* __restrict__ topic,
                            const float* __restrict__ W1s, const float* __restrict__ W2,
                            const float* __restrict__ b2,
                            float* __restrict__ out, int E) {
  int e = blockIdx.x * 8 + (threadIdx.x >> 5);
  int lane = threadIdx.x & 31;
  if (e >= E) return;
  int h = h_ids[e], r = r_ids[e], t = t_ids[e];
  float fA = isA[e], fS = aS[e], fh = topic[h], ft = topic[t];
  const float* ph = preH + (size_t)h * 256;
  const float* pr = preR + (size_t)r * 256;
  const float* pt = preT + (size_t)t * 256;
  float s = 0.f;
#pragma unroll
  for (int u = 0; u < 8; u++) {
    int c = lane + 32 * u;
    float v = qc[c] + b1[c] + ph[c] + pr[c] + pt[c]
            + fA * W1s[c] + fS * W1s[256 + c] + fh * W1s[512 + c] + ft * W1s[768 + c];
    v = fmaxf(v, 0.f);
    s += v * W2[c];
  }
#pragma unroll
  for (int off = 16; off; off >>= 1) s += __shfl_xor(s, off, 32);
  if (lane == 0) out[e] = s + b2[0];
}

// ---------------------------------------------------------------------------
// Host orchestration
// ---------------------------------------------------------------------------
static inline int cdiv(int a, int b) { return (a + b - 1) / b; }

extern "C" void kernel_launch(void* const* d_in, const int* in_sizes, int n_in,
                              void* d_out, int out_size, void* d_ws, size_t ws_size,
                              hipStream_t stream) {
  (void)n_in; (void)out_size; (void)ws_size;
  const int* h_ids = (const int*)d_in[0];
  const int* r_ids = (const int*)d_in[1];
  const int* t_ids = (const int*)d_in[2];
  const float* q_emb         = (const float*)d_in[3];
  const float* entity_embs   = (const float*)d_in[4];
  const float* relation_embs = (const float*)d_in[5];
  const int*   topic_ids     = (const int*)d_in[6];
  // d_in[7]=num_entities, d_in[8]=num_non_text (device scalars; reference constants used)
  const float* non_text_emb  = (const float*)d_in[9];
  const float* dist_emb      = (const float*)d_in[10];
  const float* W_enc     = (const float*)d_in[11];
  const float* b_enc     = (const float*)d_in[12];
  const float* gnn_W_msg = (const float*)d_in[13];
  const float* gnn_W_gate= (const float*)d_in[14];
  const float* gnn_W_upd = (const float*)d_in[15];
  const float* W_out     = (const float*)d_in[16];
  const float* scorer_W1 = (const float*)d_in[17];
  const float* scorer_b1 = (const float*)d_in[18];
  const float* scorer_W2 = (const float*)d_in[19];
  const float* scorer_b2 = (const float*)d_in[20];
  float* out = (float*)d_out;

  const int E = in_sizes[0];
  const int Ntext = in_sizes[4] / 256;
  const int R = in_sizes[5] / 256;
  const int N = 20000;                 // num_entities (reference constant)
  const int NTOP = in_sizes[6];        // 4 topic entities

  // ---- workspace layout (256B-aligned bump allocator, explicit reuse) ----
  char* wsp = (char*)d_ws;
  size_t wo = 0;
  auto wsa = [&](size_t b) -> char* {
    char* p = wsp + wo;
    wo += (b + 255) & ~(size_t)255;
    return p;
  };
  float* qe   = (float*)wsa((size_t)N * 4);
  float* qr   = (float*)wsa((size_t)R * 4);
  float* tri  = (float*)wsa((size_t)E * 4);
  int*   aidx = (int*)  wsa(32 * 4);
  float* ascr = (float*)wsa(32 * 4);
  float* topic= (float*)wsa((size_t)N * 4);
  float* degt = (float*)wsa((size_t)N * 4);
  float* degh = (float*)wsa((size_t)N * 4);
  float* sA   = (float*)wsa((size_t)N * 4);
  float* sB   = (float*)wsa((size_t)N * 4);
  float* peT  = (float*)wsa((size_t)N * 4 * 4);
  float* amask= (float*)wsa((size_t)N * 4);
  int*   dA   = (int*)  wsa((size_t)N * 4);
  int*   dB   = (int*)  wsa((size_t)N * 4);
  float* isA  = (float*)wsa((size_t)E * 4);
  float* aS   = (float*)wsa((size_t)E * 4);
  float* Xenc = (float*)wsa((size_t)N * 288 * 4);   // reused later as preH
  float* h_a  = (float*)wsa((size_t)N * 256 * 4);
  float* h_b  = (float*)wsa((size_t)N * 256 * 4);
  float* Am   = (float*)wsa((size_t)N * 256 * 4);   // reused later as h_final
  float* agg  = (float*)wsa((size_t)N * 256 * 4);   // reused later as preT
  float* gate = (float*)wsa((size_t)N * 256 * 4);
  float* Bm   = (float*)wsa((size_t)R * 256 * 4);
  float* preR = (float*)wsa((size_t)R * 256 * 4);
  float* qc   = (float*)wsa(256 * 4);
  unsigned short* WTenc = (unsigned short*)wsa((size_t)256 * 288 * 2);
  unsigned short* WTmt[2], *WTmb[2], *WTgt[2], *WTgb[2], *WTu[2];
  for (int l = 0; l < 2; l++) {
    WTmt[l] = (unsigned short*)wsa((size_t)256 * 256 * 2);
    WTmb[l] = (unsigned short*)wsa((size_t)256 * 256 * 2);
    WTgt[l] = (unsigned short*)wsa((size_t)256 * 256 * 2);
    WTgb[l] = (unsigned short*)wsa((size_t)256 * 256 * 2);
    WTu[l]  = (unsigned short*)wsa((size_t)256 * 256 * 2);
  }
  unsigned short* WTout = (unsigned short*)wsa((size_t)256 * 256 * 2);
  unsigned short* WT1h  = (unsigned short*)wsa((size_t)256 * 256 * 2);
  unsigned short* WT1r  = (unsigned short*)wsa((size_t)256 * 256 * 2);
  unsigned short* WT1t  = (unsigned short*)wsa((size_t)256 * 256 * 2);

  const dim3 blk(256);
  auto gemm = [&](const float* A, int lda, const unsigned short* BT, int K,
                  float* C, int M, int epi, const float* bias,
                  const float* ex1, const float* ex2) {
    dim3 grid(cdiv(M, 128), 2);
    hipLaunchKernelGGL(gemm_bf16_wmma, grid, blk, 0, stream,
                       A, lda, BT, K, C, M, epi, bias, ex1, ex2);
  };

  // ---- 0. weight conversion (bf16, transposed, K-padded) ----
  hipLaunchKernelGGL(wt_convert, dim3(cdiv(256 * 288, 256)), blk, 0, stream, W_enc, WTenc, 279, 288);
  for (int l = 0; l < 2; l++) {
    const float* mb = gnn_W_msg  + (size_t)l * 512 * 256;
    const float* gb = gnn_W_gate + (size_t)l * 512 * 256;
    const float* ub = gnn_W_upd  + (size_t)l * 256 * 256;
    hipLaunchKernelGGL(wt_convert, dim3(256), blk, 0, stream, mb,               WTmt[l], 256, 256);
    hipLaunchKernelGGL(wt_convert, dim3(256), blk, 0, stream, mb + 256 * 256,   WTmb[l], 256, 256);
    hipLaunchKernelGGL(wt_convert, dim3(256), blk, 0, stream, gb,               WTgt[l], 256, 256);
    hipLaunchKernelGGL(wt_convert, dim3(256), blk, 0, stream, gb + 256 * 256,   WTgb[l], 256, 256);
    hipLaunchKernelGGL(wt_convert, dim3(256), blk, 0, stream, ub,               WTu[l],  256, 256);
  }
  hipLaunchKernelGGL(wt_convert, dim3(256), blk, 0, stream, W_out,                   WTout, 256, 256);
  hipLaunchKernelGGL(wt_convert, dim3(256), blk, 0, stream, scorer_W1 + 256 * 256,   WT1h,  256, 256);
  hipLaunchKernelGGL(wt_convert, dim3(256), blk, 0, stream, scorer_W1 + 512 * 256,   WT1r,  256, 256);
  hipLaunchKernelGGL(wt_convert, dim3(256), blk, 0, stream, scorer_W1 + 768 * 256,   WT1t,  256, 256);

  // ---- 1. anchor triple selection ----
  hipLaunchKernelGGL(row_dot_q, dim3(cdiv(N, 8)), blk, 0, stream,
                     entity_embs, non_text_emb, Ntext, q_emb, qe, N);
  hipLaunchKernelGGL(row_dot_q, dim3(cdiv(R, 8)), blk, 0, stream,
                     relation_embs, relation_embs, R, q_emb, qr, R);
  hipLaunchKernelGGL(tri_score_kernel, dim3(cdiv(E, 256)), blk, 0, stream,
                     h_ids, r_ids, t_ids, qe, qr, tri, E);
  hipLaunchKernelGGL(topk32_kernel, dim3(1), dim3(1024), 0, stream, tri, E, aidx, ascr);

  // ---- 2. topic / degree / anchor scatter ----
  hipLaunchKernelGGL(zero_f32, dim3(cdiv(N, 256)), blk, 0, stream, topic, N);
  hipLaunchKernelGGL(zero_f32, dim3(cdiv(N, 256)), blk, 0, stream, degt, N);
  hipLaunchKernelGGL(zero_f32, dim3(cdiv(N, 256)), blk, 0, stream, degh, N);
  hipLaunchKernelGGL(zero_f32, dim3(cdiv(N, 256)), blk, 0, stream, amask, N);
  hipLaunchKernelGGL(zero_f32, dim3(cdiv(E, 256)), blk, 0, stream, isA, E);
  hipLaunchKernelGGL(zero_f32, dim3(cdiv(E, 256)), blk, 0, stream, aS, E);
  hipLaunchKernelGGL(topic_scatter, dim3(1), dim3(32), 0, stream, topic_ids, topic, NTOP);
  hipLaunchKernelGGL(deg_kernel, dim3(cdiv(E, 256)), blk, 0, stream, h_ids, t_ids, degh, degt, E);
  hipLaunchKernelGGL(anchor_scatter, dim3(1), dim3(32), 0, stream,
                     aidx, ascr, h_ids, t_ids, amask, isA, aS);

  // ---- 3. DDE label propagation (2 fwd + 2 rev rounds) ----
  float* sp = sA; float* sq = sB;
  hipLaunchKernelGGL(copy_f32, dim3(cdiv(N, 256)), blk, 0, stream, topic, sp, N);
  for (int rd = 0; rd < 2; rd++) {
    hipLaunchKernelGGL(zero_f32, dim3(cdiv(N, 256)), blk, 0, stream, sq, N);
    hipLaunchKernelGGL(dde_edge, dim3(cdiv(E, 256)), blk, 0, stream, h_ids, t_ids, sp, sq, E);
    hipLaunchKernelGGL(dde_div_store, dim3(cdiv(N, 256)), blk, 0, stream, sq, degt, peT, rd, N);
    float* tmp = sp; sp = sq; sq = tmp;
  }
  hipLaunchKernelGGL(copy_f32, dim3(cdiv(N, 256)), blk, 0, stream, topic, sp, N);
  for (int rd = 0; rd < 2; rd++) {
    hipLaunchKernelGGL(zero_f32, dim3(cdiv(N, 256)), blk, 0, stream, sq, N);
    hipLaunchKernelGGL(dde_edge, dim3(cdiv(E, 256)), blk, 0, stream, t_ids, h_ids, sp, sq, E);
    hipLaunchKernelGGL(dde_div_store, dim3(cdiv(N, 256)), blk, 0, stream, sq, degh, peT, 2 + rd, N);
    float* tmp = sp; sp = sq; sq = tmp;
  }

  // ---- 4. bounded BFS from anchor endpoints (3 hops, int atomicMin) ----
  int* dp = dA; int* dq = dB;
  hipLaunchKernelGGL(bfs_init, dim3(cdiv(N, 256)), blk, 0, stream, amask, dp, N);
  for (int it = 0; it < 3; it++) {
    hipLaunchKernelGGL(copy_i32, dim3(cdiv(N, 256)), blk, 0, stream, dp, dq, N);
    hipLaunchKernelGGL(bfs_edge, dim3(cdiv(E, 256)), blk, 0, stream, h_ids, t_ids, dp, dq, E);
    int* tmp = dp; dp = dq; dq = tmp;
  }

  // ---- 5. encoder:  h = relu(X @ W_enc + b_enc)   (K padded 279 -> 288) ----
  hipLaunchKernelGGL(xenc_assemble, dim3(N), dim3(288), 0, stream,
                     entity_embs, non_text_emb, Ntext, peT, dp, dist_emb, topic, amask, Xenc);
  gemm(Xenc, 288, WTenc, 288, h_a, N, 1, b_enc, nullptr, nullptr);

  // ---- 6. gated GNN, 2 layers (edge GEMM factored into N-sized GEMMs) ----
  float* hc = h_a; float* hn = h_b;
  for (int l = 0; l < 2; l++) {
    gemm(relation_embs, 256, WTmb[l], 256, Bm, R, 0, nullptr, nullptr, nullptr);
    gemm(hc, 256, WTmt[l], 256, Am, N, 0, nullptr, nullptr, nullptr);
    hipLaunchKernelGGL(zero_f32, dim3(cdiv(N * 256, 256)), blk, 0, stream, agg, N * 256);
    hipLaunchKernelGGL(edge_msg, dim3(cdiv(E, 8)), blk, 0, stream,
                       h_ids, r_ids, t_ids, Am, Bm, agg, E);
    gemm(hc,  256, WTgt[l], 256, gate, N, 0, nullptr, nullptr, nullptr);
    gemm(agg, 256, WTgb[l], 256, gate, N, 2, nullptr, nullptr, nullptr);  // += then sigmoid
    gemm(agg, 256, WTu[l],  256, hn,   N, 3, nullptr, hc, gate);          // h + gate*tanh(.)
    float* tmp = hc; hc = hn; hn = tmp;
  }

  // ---- 7. output projection + scorer pre-projections ----
  float* h_final = Am;          // Am dead after last layer
  gemm(hc, 256, WTout, 256, h_final, N, 0, nullptr, nullptr, nullptr);
  float* preH = Xenc;           // Xenc dead
  float* preT = agg;            // agg dead
  gemm(h_final, 256, WT1h, 256, preH, N, 0, nullptr, nullptr, nullptr);
  gemm(h_final, 256, WT1t, 256, preT, N, 0, nullptr, nullptr, nullptr);
  gemm(relation_embs, 256, WT1r, 256, preR, R, 0, nullptr, nullptr, nullptr);
  hipLaunchKernelGGL(q_contrib_kernel, dim3(1), blk, 0, stream, q_emb, scorer_W1, qc);

  // ---- 8. per-edge scorer ----
  hipLaunchKernelGGL(scorer_edge, dim3(cdiv(E, 8)), blk, 0, stream,
                     h_ids, r_ids, t_ids, preH, preR, preT, qc, scorer_b1,
                     isA, aS, topic, scorer_W1 + (size_t)1024 * 256, scorer_W2, scorer_b2,
                     out, E);
}
